// PAMReal_Module_30812095382087
// MI455X (gfx1250) — matrix operand order repaired and checked by reading, not compiled, not run on hardware
//
#include <hip/hip_runtime.h>
#include <hip/hip_bf16.h>
#include <stdint.h>
#include <stddef.h>

// ---------- types ----------
typedef __attribute__((ext_vector_type(16))) __bf16 v16bf;
typedef __attribute__((ext_vector_type(8)))  __bf16 v8bf;
typedef __attribute__((ext_vector_type(8)))  float  v8f;
typedef __attribute__((ext_vector_type(4)))  float  v4f;
typedef __attribute__((ext_vector_type(4)))  int    v4i;

union V16U { v16bf v; v8bf h[2]; };

#define BB 8
#define CC 256
#define NN 4096

// ---------- gfx1250 async global<->LDS paths (guarded; fallback = sync) ----------
#if defined(__AMDGCN__) && __has_builtin(__builtin_amdgcn_global_load_async_to_lds_b128)
#define HAVE_ASYNC_LD 1
#else
#define HAVE_ASYNC_LD 0
#endif
#if defined(__AMDGCN__) && __has_builtin(__builtin_amdgcn_global_store_async_from_lds_b128)
#define HAVE_ASYNC_ST 1
#else
#define HAVE_ASYNC_ST 0
#endif

#if HAVE_ASYNC_LD || HAVE_ASYNC_ST
// builtin expects int4* in AS1 (global) and AS3 (LDS); HIP prints AS1 as __device__
typedef __attribute__((address_space(1))) v4i* gas_ptr;
typedef __attribute__((address_space(3))) v4i* las_ptr;
static __device__ __forceinline__ gas_ptr to_gas(const void* p) {
  return (gas_ptr)(uintptr_t)p;
}
static __device__ __forceinline__ las_ptr to_las(const void* p) {
  // low 32 bits of a generic shared pointer are the LDS byte offset (ISA aperture map)
  return (las_ptr)(uint32_t)(uintptr_t)p;
}
#endif

static __device__ __forceinline__ void wait_async0() {
#if HAVE_ASYNC_LD
#if __has_builtin(__builtin_amdgcn_s_wait_asynccnt)
  __builtin_amdgcn_s_wait_asynccnt(0);
#else
  asm volatile("s_wait_asynccnt 0x0" ::: "memory");
#endif
#endif
}

static __device__ __forceinline__ __bf16 f2bf(float f) {
  unsigned u = __builtin_bit_cast(unsigned, f);
  unsigned r = u + 0x7FFFu + ((u >> 16) & 1u);       // round-to-nearest-even
  unsigned short s = (unsigned short)(r >> 16);
  return __builtin_bit_cast(__bf16, s);
}
static __device__ __forceinline__ float bf2f(__bf16 h) {
  unsigned u = ((unsigned)__builtin_bit_cast(unsigned short, h)) << 16;
  return __builtin_bit_cast(float, u);
}
static __device__ __forceinline__ v8f wmma_bf16(v16bf a, v16bf b, v8f c) {
  return __builtin_amdgcn_wmma_f32_16x16x32_bf16(false, a, false, b, (short)0, c,
                                                 false, false);
}

// ---------------- kernel 0: wv -> bf16 ----------------
__global__ __launch_bounds__(256) void k_cvt_wv(const float* __restrict__ wv,
                                               __bf16* __restrict__ wvbf) {
  int i = blockIdx.x * 256 + threadIdx.x;
  wvbf[i] = f2bf(wv[i]);
}

// ---------------- kernel 1: q,k,v projection ----------------
// grid: B * (N/64); block 256 (8 waves)
// outputs: qb/kb [b][n][32] bf16 (d=8 real, rest zero), vbf [b][c][n] bf16 (bias folded)
__global__ __launch_bounds__(256) void k_qkv(
    const float* __restrict__ x,  const float* __restrict__ wq, const float* __restrict__ bq,
    const float* __restrict__ wk, const float* __restrict__ bk,
    const __bf16* __restrict__ wvbf, const float* __restrict__ bv,
    __bf16* __restrict__ qb, __bf16* __restrict__ kb, __bf16* __restrict__ vbf) {
  extern __shared__ char smem[];
  __bf16* xs  = (__bf16*)smem;                  // [64][264]  x^T tile (n rows, c' cols)
  __bf16* vsm = (__bf16*)(smem + 64 * 264 * 2); // [256][72]  v tile (c rows, n cols)

  const int t  = threadIdx.x;
  const int b  = blockIdx.x >> 6;
  const int n0 = (blockIdx.x & 63) << 6;

  // ---- stage x^T tile as bf16 (coalesced global, transposed LDS store) ----
  {
    const int jj = t & 15, r0 = t >> 4;
    for (int k2 = 0; k2 < 16; ++k2) {
      const int cp = r0 + k2 * 16;
      v4f xv = *(const v4f*)(x + ((size_t)b * CC + cp) * NN + n0 + jj * 4);
#pragma unroll
      for (int i = 0; i < 4; ++i) xs[(jj * 4 + i) * 264 + cp] = f2bf(xv[i]);
    }
  }
  __syncthreads();

  // ---- q,k (K=256, D=8): VALU; all threads compute (4x redundant), group 0 stores ----
  {
    const int n = t & 63;
    float qa[8] = {0,0,0,0,0,0,0,0};
    float ka[8] = {0,0,0,0,0,0,0,0};
    for (int cp = 0; cp < CC; ++cp) {
      float xv = bf2f(xs[n * 264 + cp]);
#pragma unroll
      for (int d = 0; d < 8; ++d) {
        qa[d] += xv * wq[d * CC + cp];
        ka[d] += xv * wk[d * CC + cp];
      }
    }
    if (t < 64) {
      size_t base = ((size_t)b * NN + n0 + n) * 32;
#pragma unroll
      for (int d = 0; d < 8; ++d) {
        qb[base + d] = f2bf(qa[d] + bq[d]);
        kb[base + d] = f2bf(ka[d] + bk[d]);
      }
#pragma unroll
      for (int d = 8; d < 32; ++d) {
        qb[base + d] = f2bf(0.f);
        kb[base + d] = f2bf(0.f);
      }
    }
  }

  // ---- V = wv * x via bf16 WMMA; wave w owns c-strip [w*32, w*32+32) ----
  {
    const int w = t >> 5, L = t & 31, l16 = L & 15;
    const bool hi = (L >= 16);
    const int cw = w * 32;

    v8f zero8 = {0,0,0,0,0,0,0,0};
    v8f acc[2][4];
#pragma unroll
    for (int a2 = 0; a2 < 2; ++a2)
#pragma unroll
      for (int nt = 0; nt < 4; ++nt) acc[a2][nt] = zero8;

    for (int kk = 0; kk < 8; ++kk) {  // K-steps of 32 over c'
      V16U Af[2];
#pragma unroll
      for (int a2 = 0; a2 < 2; ++a2) {  // A = wv rows (M=c, K=c'), interleaved chunks
        const __bf16* ap =
            wvbf + ((size_t)(cw + a2 * 16 + l16)) * CC + kk * 32 + (hi ? 8 : 0);
        Af[a2].h[0] = *(const v8bf*)ap;
        Af[a2].h[1] = *(const v8bf*)(ap + 16);
      }
#pragma unroll
      for (int nt = 0; nt < 4; ++nt) {  // B = x^T (K=c', N=n), interleaved chunks
        const __bf16* bp = xs + (nt * 16 + l16) * 264 + kk * 32 + (hi ? 8 : 0);
        V16U Bf;
        Bf.h[0] = *(const v8bf*)bp;
        Bf.h[1] = *(const v8bf*)(bp + 16);
        acc[0][nt] = wmma_bf16(Af[0].v, Bf.v, acc[0][nt]);
        acc[1][nt] = wmma_bf16(Af[1].v, Bf.v, acc[1][nt]);
      }
    }

    // bias + transpose into LDS [c][n]
#pragma unroll
    for (int a2 = 0; a2 < 2; ++a2)
#pragma unroll
      for (int nt = 0; nt < 4; ++nt) {
#pragma unroll
        for (int r = 0; r < 8; ++r) {
          int c = cw + a2 * 16 + r + (hi ? 8 : 0);
          float val = acc[a2][nt][r] + bv[c];
          vsm[c * 72 + nt * 16 + l16] = f2bf(val);
        }
      }
  }
  __syncthreads();

  // ---- copy V tile to global [b][c][n] ----
  {
    const int j = t & 7, cb = t >> 3;
#pragma unroll
    for (int k = 0; k < 8; ++k) {
      int c = cb + k * 32;
#if HAVE_ASYNC_ST
      __builtin_amdgcn_global_store_async_from_lds_b128(
          to_gas(vbf + ((size_t)b * CC + c) * NN + n0 + j * 8),
          to_las(vsm + c * 72 + j * 8), 0, 0);
#else
      v8bf d8 = *(const v8bf*)(vsm + c * 72 + j * 8);
      *(v8bf*)(vbf + ((size_t)b * CC + c) * NN + n0 + j * 8) = d8;
#endif
    }
  }
  // async stores (if any) are covered by the implicit wait-idle at S_ENDPGM
}

// ---------------- kernel 2: fused attention ----------------
// grid: B * (N/64) query tiles; block 256 (8 waves). One WG owns 64 queries x 256 channels.
__global__ __launch_bounds__(256) void k_attn(
    const __bf16* __restrict__ qb, const __bf16* __restrict__ kb,
    const __bf16* __restrict__ vbf, const float* __restrict__ x,
    const float* __restrict__ gamma, float* __restrict__ out) {
  extern __shared__ char smem[];
  __bf16* vsm  = (__bf16*)smem;                     // [2][256][72] V tiles (double buf)
  __bf16* ps   = (__bf16*)(smem + 73728);           // [2][64][72]  P tiles   (double buf)
  __bf16* qs   = (__bf16*)(smem + 92160);           // [64][32]     Q tile
  float*  den4 = (float*)(smem + 96256);            // [64][4] partial denominators
  float*  dfin = (float*)(smem + 97280);            // [64]    denominators
  float*  os   = (float*)smem;                      // overlay: [256][68] O staging

  const int t  = threadIdx.x;
  const int b  = blockIdx.x >> 6;
  const int m0 = (blockIdx.x & 63) << 6;
  const int w = t >> 5, L = t & 31, l16 = L & 15;
  const bool hi = (L >= 16);
  const int cw = w * 32;
  const __bf16* vbase = vbf + (size_t)b * CC * NN;  // [c][n] for this batch

  // per-thread V-staging coordinates
  const int vj = t & 7, vcb = t >> 3;

  // ---- stage Q tile (64 x 32 bf16) ----
  {
    const int row = t >> 2, ch = t & 3;
    *(v8bf*)(qs + row * 32 + ch * 8) =
        *(const v8bf*)(qb + ((size_t)b * NN + m0 + row) * 32 + ch * 8);
  }
  __syncthreads();

  // wave tile assignment for S^T (16 tiles of (tn x tm), 2 per wave)
  const int tt0 = w * 2;
  int tmA[2], tnA[2];
  tmA[0] = tt0 >> 2;       tnA[0] = tt0 & 3;
  tmA[1] = (tt0 + 1) >> 2; tnA[1] = (tt0 + 1) & 3;

  // preload Q B-fragments (N=m, K=d; d padded to 32)
  v16bf Qf[2];
#pragma unroll
  for (int i = 0; i < 2; ++i) {
    V16U u;
    const __bf16* qp = qs + (tmA[i] * 16 + l16) * 32 + (hi ? 8 : 0);
    u.h[0] = *(const v8bf*)qp;
    u.h[1] = *(const v8bf*)(qp + 16);
    Qf[i] = u.v;
  }

  v8f zero8 = {0,0,0,0,0,0,0,0};
  v8f Oacc[4][2];
#pragma unroll
  for (int tm = 0; tm < 4; ++tm)
#pragma unroll
    for (int tc = 0; tc < 2; ++tc) Oacc[tm][tc] = zero8;
  float dacc[2] = {0.f, 0.f};

#if HAVE_ASYNC_LD
  // prologue: async-stage V tile 0 into buffer 0
#pragma unroll
  for (int k = 0; k < 8; ++k) {
    int c = vcb + k * 32;
    __builtin_amdgcn_global_load_async_to_lds_b128(
        to_gas(vbase + (size_t)c * NN + 0 + vj * 8),
        to_las(vsm + c * 72 + vj * 8), 0, 0);
  }
#endif

  // ---- stream over all 64 key tiles ----
  for (int it = 0; it < 64; ++it) {
    const int nb  = it * 64;
    const int buf = it & 1;
    __bf16* vbuf = vsm + buf * (256 * 72);
    __bf16* pbuf = ps + buf * (64 * 72);

#if !HAVE_ASYNC_LD
    // sync fallback: stage V tile [c][n] -> LDS (coalesced b128)
    {
#pragma unroll
      for (int k = 0; k < 8; ++k) {
        int c = vcb + k * 32;
        v8bf d8 = *(const v8bf*)(vbase + (size_t)c * NN + nb + vj * 8);
        *(v8bf*)(vbuf + c * 72 + vj * 8) = d8;
      }
    }
#endif

    // prefetch next iteration's K rows into cache
    if (it + 1 < 64) {
      const int nb2 = nb + 64;
#pragma unroll
      for (int i = 0; i < 2; ++i)
        __builtin_prefetch(kb + ((size_t)b * NN + nb2 + tnA[i] * 16 + l16) * 32, 0, 1);
    }

    // S^T = K * Q^T (one WMMA per tile, K=32), exp, P^T -> ps, running denom
#pragma unroll
    for (int i = 0; i < 2; ++i) {
      V16U Ka;  // A = K rows (M=n, K=d), interleaved chunks
      const __bf16* kp =
          kb + ((size_t)b * NN + nb + tnA[i] * 16 + l16) * 32 + (hi ? 8 : 0);
      Ka.h[0] = *(const v8bf*)kp;
      Ka.h[1] = *(const v8bf*)(kp + 16);
      v8f s = wmma_bf16(Ka.v, Qf[i], zero8);
      v8bf e8;
      float ls = 0.f;
#pragma unroll
      for (int r = 0; r < 8; ++r) {
        float e = __expf(s[r]);  // energies small enough for fp32, no max pass
        ls += e;
        e8[r] = f2bf(e);
      }
      dacc[i] += ls;
      // transpose-trick: lane's fixed index m is the row, n contiguous -> 1 b128
      *(v8bf*)(pbuf + (tmA[i] * 16 + l16) * 72 + tnA[i] * 16 + (hi ? 8 : 0)) = e8;
    }

    wait_async0();    // own async V[it] writes to LDS complete
    __syncthreads();  // single barrier per iteration (double buffered V and P)

#if HAVE_ASYNC_LD
    // software pipeline: async-stage V[it+1] into the other buffer; all waves are
    // provably past their reads of that buffer (they happened before this barrier).
    if (it + 1 < 64) {
      __bf16* vnext = vsm + (buf ^ 1) * (256 * 72);
      const int nb2 = nb + 64;
#pragma unroll
      for (int k = 0; k < 8; ++k) {
        int c = vcb + k * 32;
        __builtin_amdgcn_global_load_async_to_lds_b128(
            to_gas(vbase + (size_t)c * NN + nb2 + vj * 8),
            to_las(vnext + c * 72 + vj * 8), 0, 0);
      }
    }
#endif

    // O += P * V  (A = ps rows m, B = V [c][n] straight-K)
#pragma unroll
    for (int kkI = 0; kkI < 2; ++kkI) {
      const int kk = kkI * 32;
      V16U Bf[2];
#pragma unroll
      for (int tc = 0; tc < 2; ++tc) {
        const __bf16* vp = vbuf + (cw + tc * 16 + l16) * 72 + kk + (hi ? 8 : 0);
        Bf[tc].h[0] = *(const v8bf*)vp;
        Bf[tc].h[1] = *(const v8bf*)(vp + 16);
      }
#pragma unroll
      for (int tm = 0; tm < 4; ++tm) {
        V16U Pa;
        const __bf16* pp = pbuf + (tm * 16 + l16) * 72 + kk + (hi ? 8 : 0);
        Pa.h[0] = *(const v8bf*)pp;
        Pa.h[1] = *(const v8bf*)(pp + 16);
        Oacc[tm][0] = wmma_bf16(Pa.v, Bf[0].v, Oacc[tm][0]);
        Oacc[tm][1] = wmma_bf16(Pa.v, Bf[1].v, Oacc[tm][1]);
      }
    }
  }

  // ---- denominators: pair-sum across lane halves, combine over tn ----
#pragma unroll
  for (int i = 0; i < 2; ++i) {
    float tot = dacc[i] + __shfl_xor(dacc[i], 16, 32);
    if (!hi) den4[(tmA[i] * 16 + l16) * 4 + tnA[i]] = tot;
  }
  __syncthreads();  // den4 ready AND all V-tile reads complete (os may overlay vsm)
  if (t < 64)
    dfin[t] = den4[t * 4 + 0] + den4[t * 4 + 1] + den4[t * 4 + 2] + den4[t * 4 + 3];

  // stage O to LDS [c][m] (transpose-trick: m contiguous per lane)
#pragma unroll
  for (int tm = 0; tm < 4; ++tm)
#pragma unroll
    for (int tc = 0; tc < 2; ++tc) {
      float* op = os + (cw + tc * 16 + l16) * 68 + tm * 16 + (hi ? 8 : 0);
      v8f a = Oacc[tm][tc];
      *(v4f*)op       = (v4f){a[0], a[1], a[2], a[3]};
      *(v4f*)(op + 4) = (v4f){a[4], a[5], a[6], a[7]};
    }
  __syncthreads();

  // ---- final: out = gamma * O/denom + x, coalesced b128 ----
  {
    const float g = gamma[0];
    const int j = t & 7, cb = t >> 3;
#pragma unroll
    for (int k = 0; k < 8; ++k) {
      int c = cb + k * 32;
#pragma unroll
      for (int h2 = 0; h2 < 2; ++h2) {
        v4f o  = *(const v4f*)(os + c * 68 + j * 8 + h2 * 4);
        v4f dv = *(const v4f*)(dfin + j * 8 + h2 * 4);
        size_t gi = ((size_t)b * CC + c) * NN + m0 + j * 8 + h2 * 4;
        v4f xv = *(const v4f*)(x + gi);
        v4f res = g * (o / dv) + xv;
        *(v4f*)(out + gi) = res;
      }
    }
  }
}

// ---------------- launch ----------------
extern "C" void kernel_launch(void* const* d_in, const int* in_sizes, int n_in,
                              void* d_out, int out_size, void* d_ws, size_t ws_size,
                              hipStream_t stream) {
  const float* x     = (const float*)d_in[0];
  const float* wq    = (const float*)d_in[1];
  const float* bq    = (const float*)d_in[2];
  const float* wk    = (const float*)d_in[3];
  const float* bk    = (const float*)d_in[4];
  const float* wv    = (const float*)d_in[5];
  const float* bv    = (const float*)d_in[6];
  const float* gamma = (const float*)d_in[7];
  float* out = (float*)d_out;

  char* ws = (char*)d_ws;
  __bf16* wvbf = (__bf16*)(ws);                              // 256*256*2      = 128 KB
  __bf16* qb   = (__bf16*)(ws + 131072);                     // 8*4096*32*2    = 2 MB
  __bf16* kb   = (__bf16*)(ws + 131072 + 2097152);           // 2 MB
  __bf16* vbf  = (__bf16*)(ws + 131072 + 2 * 2097152);       // 8*256*4096*2   = 16 MB

  k_cvt_wv<<<dim3(256), dim3(256), 0, stream>>>(wv, wvbf);
  k_qkv<<<dim3(BB * (NN / 64)), dim3(256), 70656, stream>>>(
      x, wq, bq, wk, bk, wvbf, bv, qb, kb, vbf);
  k_attn<<<dim3(BB * (NN / 64)), dim3(256), 97536, stream>>>(
      qb, kb, vbf, x, gamma, out);
}
